// Transformer2D_64768106824183
// MI455X (gfx1250) — compile-verified, matrix operand order checked by reading, not run on hardware
//
#include <hip/hip_runtime.h>

// ---------------- WMMA fragment types ----------------
typedef __attribute__((ext_vector_type(16))) _Float16 v16h;
typedef __attribute__((ext_vector_type(8)))  _Float16 v8h;
typedef __attribute__((ext_vector_type(8)))  float    v8f;

// ---------------- Problem constants ----------------
#define DDIM   256
#define SVIEW  8
#define DHH    32
#define HIDD   1024
#define TM     16          // tokens per workgroup
#define NTHR   512         // 16 wave32 per workgroup
#define LDH    264         // f16 row stride (256 + 8 pad halves = 528B, 16B aligned)
#define LDF1   1032        // f16 row stride for 1024-wide hidden

// ws layout (in _Float16 elements), all 16B aligned
#define WQT   0u
#define WKT   65536u
#define WVT   131072u
#define WOT   196608u
#define WA1T  262144u       // [32][256]
#define WA2T  270336u       // [256][32]
#define WP2T  278528u       // [256][32]
#define WF1T  286720u       // [1024][256]
#define WF2T  548864u       // [256][1024]

// LDS offsets (bytes)
#define SM_O     0u         // p3: async f32 k-stage [16][256] | p4/5: sO f32[16][256]
#define SM_QP    16384u     // p2-3: sQP f16[16][LDH] | p5: sY
#define SM_X     24832u     // p1-2: sX | p3: sK | p4: sXO16 | p5: sF1 (33024B)
#define SM_ATT   33280u     // p3: sATT f16[16][LDH]
#define SM_PH    41728u     // p3: sPH f16[16][32]
#define SM_H     42752u     // p3: sH  f16[16][32]
#define SM_XO    43776u     // p3-4: sXO f32[16][256]
#define SM_TOTAL 60160u

__device__ __forceinline__ v8f v8f_splat(float x) {
  v8f r;
#pragma unroll
  for (int i = 0; i < 8; ++i) r[i] = x;
  return r;
}

// 16x32 f16 A-fragment (or B-fragment from [n][k] transposed weights).
// lane holds row (lane&15); kh = (lane>>4)*8; two contiguous 16B loads.
__device__ __forceinline__ v16h ld_frag16(const _Float16* __restrict__ base,
                                          int strideH, int k0, int lane) {
  const int m  = lane & 15;
  const int kh = (lane >> 4) << 3;
  const _Float16* p = base + m * strideH + k0 + kh;
  v8h lo = *reinterpret_cast<const v8h*>(p);
  v8h hi = *reinterpret_cast<const v8h*>(p + 16);
  v16h r;
#pragma unroll
  for (int i = 0; i < 8; ++i) { r[i] = lo[i]; r[i + 8] = hi[i]; }
  return r;
}

__device__ __forceinline__ v8f wmma_acc(v16h a, v16h b, v8f c) {
  return __builtin_amdgcn_wmma_f32_16x16x32_f16(false, a, false, b, (short)0, c,
                                                false, false);
}

__device__ __forceinline__ v8f wmma_gemm(const _Float16* __restrict__ aBase, int aStride,
                                         const _Float16* __restrict__ bBase, int bStride,
                                         int K, int lane, v8f acc) {
#pragma unroll 4
  for (int k0 = 0; k0 < K; k0 += 32) {
    v16h a = ld_frag16(aBase, aStride, k0, lane);
    v16h b = ld_frag16(bBase, bStride, k0, lane);
    acc = wmma_acc(a, b, acc);
  }
  return acc;
}

__device__ __forceinline__ float wave_sum(float s) {
#pragma unroll
  for (int off = 16; off; off >>= 1) s += __shfl_xor(s, off, 32);
  return s;
}

// ---- CDNA5 async global->LDS copy (ASYNCcnt-tracked), 16B per lane ----
__device__ __forceinline__ void async_copy16(unsigned lds_addr, const void* gaddr) {
  unsigned long long ga = (unsigned long long)(uintptr_t)gaddr;
  asm volatile("global_load_async_to_lds_b128 %0, %1, off"
               :: "v"(lds_addr), "v"(ga)
               : "memory");
}
__device__ __forceinline__ void wait_async0() {
  asm volatile("s_wait_asynccnt 0x0" ::: "memory");
}

// Stage one 16x256 f32 k tile (16KB = 1024 x 16B chunks) into LDS @ SM_O
__device__ __forceinline__ void stage_k_async(const float* __restrict__ kbase,
                                              char* smem, int tid) {
  unsigned lds0 = (unsigned)(uintptr_t)(smem + SM_O);
#pragma unroll
  for (int j = 0; j < 2; ++j) {
    int c = tid + NTHR * j;                 // chunk 0..1023
    async_copy16(lds0 + c * 16, (const void*)(kbase + c * 4));
  }
}

// ---------------- weight prep: f32 [K][N] -> f16 [N][K] ----------------
__global__ void wprep_tr_f16(const float* __restrict__ src, _Float16* __restrict__ dst,
                             int K, int N) {
  int idx = blockIdx.x * blockDim.x + threadIdx.x;
  if (idx >= K * N) return;
  int n = idx / K, kk = idx - n * K;
  dst[idx] = (_Float16)src[(size_t)kk * N + n];
}

// ---------------- fused transformer block ----------------
__launch_bounds__(NTHR, 1)
__global__ void t2d_fused_kernel(
    const float* __restrict__ q, const float* __restrict__ k,
    const float* __restrict__ pos,
    const float* __restrict__ posW1, const float* __restrict__ posB1,
    const float* __restrict__ posB2,
    const float* __restrict__ attnB1, const float* __restrict__ attnB2,
    const float* __restrict__ outB,
    const float* __restrict__ ln1g, const float* __restrict__ ln1b,
    const float* __restrict__ ln2g, const float* __restrict__ ln2b,
    const float* __restrict__ ffB1, const float* __restrict__ ffB2,
    const _Float16* __restrict__ ws, float* __restrict__ out) {
  __shared__ __align__(16) char smem[SM_TOTAL];
  const int tid  = threadIdx.x;
  const int lane = tid & 31;
  const int wv   = tid >> 5;             // 0..15, also this wave's N-tile
  const int T0   = blockIdx.x * TM;
  const int nlo  = lane & 15;
  const int hi   = lane >> 4;
  const int mb   = hi << 3;

  // Kick off async stage of row-tile 0 of k immediately (lands in SM_O region).
  stage_k_async(k + (size_t)(T0 * SVIEW) * DDIM, smem, tid);

  _Float16* sQP = (_Float16*)(smem + SM_QP);
  float*    sXO = (float*)(smem + SM_XO);

  // ---- Phase 1: LayerNorm(q) -> sX (f16); one row per wave ----
  {
    _Float16* sX = (_Float16*)(smem + SM_X);
    const int rr = wv;
    const float* qr = q + (size_t)(T0 + rr) * DDIM;
    float v[8], s = 0.f;
#pragma unroll
    for (int i = 0; i < 8; ++i) { v[i] = qr[lane + 32 * i]; s += v[i]; }
    float mu = wave_sum(s) * (1.f / 256.f);
    float vs = 0.f;
#pragma unroll
    for (int i = 0; i < 8; ++i) { float d = v[i] - mu; vs += d * d; }
    float rstd = rsqrtf(wave_sum(vs) * (1.f / 256.f) + 1e-6f);
#pragma unroll
    for (int i = 0; i < 8; ++i) {
      int c = lane + 32 * i;
      sX[rr * LDH + c] = (_Float16)((v[i] - mu) * rstd * ln1g[c] + ln1b[c]);
    }
  }
  __syncthreads();

  // ---- Phase 2: QP = X @ qW (no bias) -> sQP ----
  {
    const _Float16* sX = (const _Float16*)(smem + SM_X);
    v8f acc = v8f_splat(0.f);
    acc = wmma_gemm(sX, LDH, ws + WQT + wv * 16 * DDIM, DDIM, DDIM, lane, acc);
#pragma unroll
    for (int j = 0; j < 8; ++j)
      sQP[(mb + j) * LDH + wv * 16 + nlo] = (_Float16)acc[j];
  }

  // Hoist this wave's kW/vW/posW2/attnW2 B-fragments: loaded ONCE, reused for
  // all 8 row-tiles (cuts kp/vp weight L2 traffic 8x). 16 v16h = 128 VGPRs.
  v16h bK[8], bV[8];
#pragma unroll
  for (int i = 0; i < 8; ++i) {
    bK[i] = ld_frag16(ws + WKT + wv * 16 * DDIM, DDIM, i * 32, lane);
    bV[i] = ld_frag16(ws + WVT + wv * 16 * DDIM, DDIM, i * 32, lane);
  }
  v16h bP2 = ld_frag16(ws + WP2T + wv * 16 * DHH, DHH, 0, lane);
  v16h bA2 = ld_frag16(ws + WA2T + wv * 16 * DHH, DHH, 0, lane);
  const int nW  = wv * 16 + nlo;
  const float pb2 = posB2[nW];
  const float ab2 = attnB2[nW];
  __syncthreads();

  // ---- Phase 3: attention, 8 row-tiles (16 rows = 2 tokens x 8 views) ----
  for (int rt = 0; rt < 8; ++rt) {
    _Float16* sK   = (_Float16*)(smem + SM_X);
    _Float16* sATT = (_Float16*)(smem + SM_ATT);
    _Float16* sPH  = (_Float16*)(smem + SM_PH);
    _Float16* sH   = (_Float16*)(smem + SM_H);
    const int R0 = rt * 16;

    wait_async0();                 // k tile rt has landed in SM_O
    __syncthreads();

    // f32 stage -> f16 sK ; pos-MLP hidden (4 -> 32, relu), one elem/thread
    {
      const float* kst = (const float*)(smem + SM_O);
      for (int idx = tid; idx < 16 * DDIM; idx += NTHR)
        sK[(idx >> 8) * LDH + (idx & 255)] = (_Float16)kst[idx];
      int lr = tid >> 5, hc = tid & 31;    // 512 = 16x32 exactly
      const float* pr = pos + (size_t)(T0 * SVIEW + R0 + lr) * 4;
      float a = posB1[hc];
#pragma unroll
      for (int p = 0; p < 4; ++p) a += pr[p] * posW1[p * DHH + hc];
      sPH[lr * DHH + hc] = (_Float16)fmaxf(a, 0.f);
    }
    __syncthreads();

    // prefetch next k tile via async DMA while we compute this one
    if (rt < 7)
      stage_k_async(k + (size_t)(T0 * SVIEW + R0 + 16) * DDIM, smem, tid);

    // PE = posB2 + PH @ posW2 (C-operand fusion); kp/vp on hoisted B-frags
    v8f pe = v8f_splat(pb2);
    {
      v16h a = ld_frag16(sPH, DHH, 0, lane);
      pe = wmma_acc(a, bP2, pe);
    }
    v8f kp = pe, vp = pe;
#pragma unroll
    for (int i = 0; i < 8; ++i) {
      v16h a = ld_frag16(sK, LDH, i * 32, lane);
      kp = wmma_acc(a, bK[i], kp);
      vp = wmma_acc(a, bV[i], vp);
    }
    // attn input = kp - qp(token) + pe : token = 2*rt + hi (constant per lane)
    float qpv = (float)sQP[(2 * rt + hi) * LDH + nW];
#pragma unroll
    for (int j = 0; j < 8; ++j)
      sATT[(mb + j) * LDH + nW] = (_Float16)(kp[j] - qpv);
    __syncthreads();

    // H = relu(ATT @ attnW1 + b1) : 16x32, waves 0,1
    if (wv < 2) {
      int n = wv * 16 + nlo;
      v8f h = v8f_splat(attnB1[n]);
      h = wmma_gemm(sATT, LDH, ws + WA1T + wv * 16 * DDIM, DDIM, DDIM, lane, h);
#pragma unroll
      for (int j = 0; j < 8; ++j)
        sH[(mb + j) * DHH + n] = (_Float16)fmaxf(h[j], 0.f);
    }
    __syncthreads();

    // A = H @ attnW2 + b2 ; softmax over S = the 8 acc VGPRs of each lane
    {
      v8f a = v8f_splat(ab2);
      v16h af = ld_frag16(sH, DHH, 0, lane);
      a = wmma_acc(af, bA2, a);
      float mx = a[0];
#pragma unroll
      for (int j = 1; j < 8; ++j) mx = fmaxf(mx, a[j]);
      float e[8], ssum = 0.f;
#pragma unroll
      for (int j = 0; j < 8; ++j) { e[j] = __expf(a[j] - mx); ssum += e[j]; }
      float xv = 0.f;
#pragma unroll
      for (int j = 0; j < 8; ++j) xv += vp[j] * e[j];   // vp already has +pe
      sXO[(2 * rt + hi) * DDIM + nW] = xv / ssum;
    }
    __syncthreads();
  }

  // ---- Phase 4: O = XO @ outW + outB + q(residual) -> sO (f32) ----
  {
    _Float16* sXO16 = (_Float16*)(smem + SM_X);
    float*    sO    = (float*)(smem + SM_O);
    for (int idx = tid; idx < 16 * DDIM; idx += NTHR)
      sXO16[(idx >> 8) * LDH + (idx & 255)] = (_Float16)sXO[idx];
    __syncthreads();
    v8f acc = v8f_splat(outB[nW]);
    acc = wmma_gemm(sXO16, LDH, ws + WOT + wv * 16 * DDIM, DDIM, DDIM, lane, acc);
#pragma unroll
    for (int j = 0; j < 8; ++j) {
      int m = mb + j;
      sO[m * DDIM + nW] = acc[j] + q[(size_t)(T0 + m) * DDIM + nW];
    }
    __syncthreads();
  }

  // ---- Phase 5: FFN block with LN2 + residual ----
  {
    const float* sO = (const float*)(smem + SM_O);
    _Float16*    sY = (_Float16*)(smem + SM_QP);
    _Float16*   sF1 = (_Float16*)(smem + SM_X);
    {
      const int rr = wv;
      float v[8], s = 0.f;
#pragma unroll
      for (int i = 0; i < 8; ++i) { v[i] = sO[rr * DDIM + lane + 32 * i]; s += v[i]; }
      float mu = wave_sum(s) * (1.f / 256.f);
      float vs = 0.f;
#pragma unroll
      for (int i = 0; i < 8; ++i) { float d = v[i] - mu; vs += d * d; }
      float rstd = rsqrtf(wave_sum(vs) * (1.f / 256.f) + 1e-6f);
#pragma unroll
      for (int i = 0; i < 8; ++i) {
        int c = lane + 32 * i;
        sY[rr * LDH + c] = (_Float16)((v[i] - mu) * rstd * ln2g[c] + ln2b[c]);
      }
    }
    __syncthreads();
    // F1 = relu(Y @ ffW1 + b1) : 16x1024, 64 N-tiles / 16 waves
#pragma unroll
    for (int i = 0; i < 4; ++i) {
      int nt = wv + i * 16;
      int n  = nt * 16 + nlo;
      v8f acc = v8f_splat(ffB1[n]);
      acc = wmma_gemm(sY, LDH, ws + WF1T + nt * 16 * DDIM, DDIM, DDIM, lane, acc);
#pragma unroll
      for (int j = 0; j < 8; ++j)
        sF1[(mb + j) * LDF1 + n] = (_Float16)fmaxf(acc[j], 0.f);
    }
    __syncthreads();
    // out = F1 @ ffW2 + b2 + O  (K = 1024)
    {
      v8f acc = v8f_splat(ffB2[nW]);
      acc = wmma_gemm(sF1, LDF1, ws + WF2T + wv * 16 * HIDD, HIDD, HIDD, lane, acc);
#pragma unroll
      for (int j = 0; j < 8; ++j) {
        int m = mb + j;
        out[(size_t)(T0 + m) * DDIM + nW] = acc[j] + sO[m * DDIM + nW];
      }
    }
  }
}

extern "C" void kernel_launch(void* const* d_in, const int* in_sizes, int n_in,
                              void* d_out, int out_size, void* d_ws, size_t ws_size,
                              hipStream_t stream) {
  (void)in_sizes; (void)n_in; (void)out_size; (void)ws_size;
  const float* q      = (const float*)d_in[0];
  const float* k      = (const float*)d_in[1];
  const float* pos    = (const float*)d_in[2];
  const float* qW     = (const float*)d_in[3];
  const float* kW     = (const float*)d_in[4];
  const float* vW     = (const float*)d_in[5];
  const float* posW1  = (const float*)d_in[6];
  const float* posB1  = (const float*)d_in[7];
  const float* posW2  = (const float*)d_in[8];
  const float* posB2  = (const float*)d_in[9];
  const float* attnW1 = (const float*)d_in[10];
  const float* attnB1 = (const float*)d_in[11];
  const float* attnW2 = (const float*)d_in[12];
  const float* attnB2 = (const float*)d_in[13];
  const float* outW   = (const float*)d_in[14];
  const float* outB   = (const float*)d_in[15];
  const float* ln1g   = (const float*)d_in[16];
  const float* ln1b   = (const float*)d_in[17];
  const float* ln2g   = (const float*)d_in[18];
  const float* ln2b   = (const float*)d_in[19];
  const float* ffW1   = (const float*)d_in[20];
  const float* ffB1   = (const float*)d_in[21];
  const float* ffW2   = (const float*)d_in[22];
  const float* ffB2   = (const float*)d_in[23];
  _Float16* ws = (_Float16*)d_ws;

  auto tr = [&](const float* src, unsigned off, int K, int N) {
    int total = K * N;
    wprep_tr_f16<<<(total + 255) / 256, 256, 0, stream>>>(src, ws + off, K, N);
  };
  tr(qW,     WQT,  256, 256);
  tr(kW,     WKT,  256, 256);
  tr(vW,     WVT,  256, 256);
  tr(outW,   WOT,  256, 256);
  tr(attnW1, WA1T, 256, 32);
  tr(attnW2, WA2T, 32,  256);
  tr(posW2,  WP2T, 32,  256);
  tr(ffW1,   WF1T, 256, 1024);
  tr(ffW2,   WF2T, 1024, 256);

  // 512*64 tokens / 16 tokens per block = 2048 blocks of 16 wave32
  t2d_fused_kernel<<<2048, NTHR, 0, stream>>>(
      q, k, pos, posW1, posB1, posB2, attnB1, attnB2, outB,
      ln1g, ln1b, ln2g, ln2b, ffB1, ffB2, ws, (float*)d_out);
}
